// AutoencoderVQ_63496796504187
// MI455X (gfx1250) — matrix operand order repaired and checked by reading, not compiled, not run on hardware
//
#include <hip/hip_runtime.h>

typedef __attribute__((ext_vector_type(16))) __bf16 v16bf;
typedef __attribute__((ext_vector_type(8)))  __bf16 v8bf;
typedef __attribute__((ext_vector_type(8)))  float  v8f;

#define B_DIM   8
#define C_DIM   256
#define HW      4096
#define E_DIM   32
#define K_DIM   8192
#define N_PIX   32768

// search-kernel staging geometry
#define CHUNK_CODES 128                    // codes per async stage
#define CHUNK_BYTES (CHUNK_CODES * 64)     // 8 KB source bytes per stage
#define ROW_PITCH   80                     // padded LDS row pitch (bytes): conflict-free b128 reads
#define BUF_BYTES   (CHUNK_CODES * ROW_PITCH)  // 10240 B per buffer
#define N_STAGES    (K_DIM / CHUNK_CODES)  // 64

// ---------------------------------------------------------------------------
// Prep: L2-normalize embedding rows -> bf16 [K,32]. One wave (32 lanes) / row.
// ---------------------------------------------------------------------------
__global__ __launch_bounds__(256) void vq_prep_norm(const float* __restrict__ emb,
                                                    __bf16* __restrict__ enb) {
    const int warp = threadIdx.x >> 5, lane = threadIdx.x & 31;
    const int k = blockIdx.x * 8 + warp;
    float v = emb[(size_t)k * E_DIM + lane];
    float s = v * v;
    #pragma unroll
    for (int off = 16; off; off >>= 1) s += __shfl_xor(s, off, 32);
    s = fmaxf(sqrtf(s), 1e-12f);
    enb[(size_t)k * E_DIM + lane] = (__bf16)(v / s);
}

// Prep: Wq f32 [E,C] -> bf16 (same layout)
__global__ __launch_bounds__(256) void vq_prep_wq(const float* __restrict__ wq,
                                                  __bf16* __restrict__ wqb) {
    int i = blockIdx.x * 256 + threadIdx.x;   // 8192 elements
    wqb[i] = (__bf16)wq[i];
}

// ---------------------------------------------------------------------------
// 1x1 conv via bf16 WMMA: z[n,e] = sum_c x[n,c]*Wq[e,c] + bq[e]
// Block = 128 thr (4 waves) = 64 pixels; x tile staged in LDS as bf16.
// ---------------------------------------------------------------------------
__global__ __launch_bounds__(128) void vq_conv(const float* __restrict__ x,
                                               const float* __restrict__ bq,
                                               const __bf16* __restrict__ wqb,
                                               float* __restrict__ zf,
                                               __bf16* __restrict__ zb) {
    __shared__ __bf16 xt[64][264];            // pad: stride 132 dwords -> banks spread
    const int P0 = blockIdx.x * 64;
    const int b  = P0 >> 12;
    const int p0 = P0 & (HW - 1);
    const float* xb = x + (size_t)b * C_DIM * HW + p0;
    for (int i = threadIdx.x; i < 64 * C_DIM; i += 128) {
        int c = i >> 6, t = i & 63;
        xt[t][c] = (__bf16)xb[(size_t)c * HW + t];
    }
    __syncthreads();

    const int wave = threadIdx.x >> 5, lane = threadIdx.x & 31;
    const int hi = lane >> 4, lo = lane & 15;
    const int arow = wave * 16 + lo;          // A-matrix row M = lane&15

    v8f acc0, acc1;
    const float b0 = bq[lo], b1 = bq[16 + lo];
    #pragma unroll
    for (int j = 0; j < 8; ++j) { acc0[j] = b0; acc1[j] = b1; }

    #pragma unroll
    for (int kc = 0; kc < 8; ++kc) {          // K = 256 = 8 x 32
        v16bf a;
        v8bf alo = *(const v8bf*)&xt[arow][kc * 32 + hi * 16];
        v8bf ahi = *(const v8bf*)&xt[arow][kc * 32 + hi * 16 + 8];
        #pragma unroll
        for (int i = 0; i < 8; ++i) { a[i] = alo[i]; a[i + 8] = ahi[i]; }
        const int kb = kc * 32 + hi * 16;     // B-matrix K base for this half-wave
        {
            const __bf16* wp = wqb + (size_t)lo * C_DIM + kb;
            v8bf blo = *(const v8bf*)wp, bh = *(const v8bf*)(wp + 8);
            v16bf bm;
            #pragma unroll
            for (int i = 0; i < 8; ++i) { bm[i] = blo[i]; bm[i + 8] = bh[i]; }
            acc0 = __builtin_amdgcn_wmma_f32_16x16x32_bf16(false, a, false, bm,
                                                           (short)0, acc0, false, false);
        }
        {
            const __bf16* wp = wqb + (size_t)(16 + lo) * C_DIM + kb;
            v8bf blo = *(const v8bf*)wp, bh = *(const v8bf*)(wp + 8);
            v16bf bm;
            #pragma unroll
            for (int i = 0; i < 8; ++i) { bm[i] = blo[i]; bm[i + 8] = bh[i]; }
            acc1 = __builtin_amdgcn_wmma_f32_16x16x32_bf16(false, a, false, bm,
                                                           (short)0, acc1, false, false);
        }
    }
    #pragma unroll
    for (int j = 0; j < 8; ++j) {             // C row M = j + 8*hi
        int pix = P0 + wave * 16 + j + 8 * hi;
        size_t o = (size_t)pix * E_DIM;
        zf[o + lo]      = acc0[j];  zb[o + lo]      = (__bf16)acc0[j];
        zf[o + 16 + lo] = acc1[j];  zb[o + 16 + lo] = (__bf16)acc1[j];
    }
}

// ---------------------------------------------------------------------------
// Code search: fused GEMM + argmax. 256 threads = 8 waves x 16 pixels.
// Embedding is double-buffered into LDS with GLOBAL_LOAD_ASYNC_TO_LDS_B128
// (ASYNCcnt), shared by all 8 waves. Each 16x16 score tile is exactly one
// v_wmma_f32_16x16x32_bf16 (K = E = 32).
// ---------------------------------------------------------------------------
__global__ __launch_bounds__(256) void vq_search(const __bf16* __restrict__ zb,
                                                 const __bf16* __restrict__ enb,
                                                 int* __restrict__ idxout) {
    __shared__ __align__(16) char ebuf[2][BUF_BYTES];   // 2 x 10 KB padded stage buffers

    const int tid  = threadIdx.x;
    const int wave = tid >> 5, lane = tid & 31;
    const int hi = lane >> 4, lo = lane & 15;
    const int pixbase = blockIdx.x * 128 + wave * 16;

    // LDS byte offset = low 32 bits of the generic shared-memory address
    const uint32_t lds0 = (uint32_t)(uintptr_t)&ebuf[0][0];
    const uint64_t gbase = (uint64_t)(uintptr_t)enb;

    // A fragment: loop-invariant z rows for this wave's 16 pixels
    v16bf a;
    {
        const __bf16* zr = zb + (size_t)(pixbase + lo) * E_DIM;
        v8bf alo = *(const v8bf*)(zr + hi * 16);
        v8bf ahi = *(const v8bf*)(zr + hi * 16 + 8);
        #pragma unroll
        for (int i = 0; i < 8; ++i) { a[i] = alo[i]; a[i + 8] = ahi[i]; }
    }

    // loop-invariant zero accumulator block (never overwritten -> no per-tile movs)
    const v8f zacc = {0.f, 0.f, 0.f, 0.f, 0.f, 0.f, 0.f, 0.f};

    float best[8]; int bidx[8];
    #pragma unroll
    for (int j = 0; j < 8; ++j) { best[j] = -3.402823466e38f; bidx[j] = 0; }

    // per-thread async stage: 2 x 16B. row = tid/4 (and +64), padded pitch 80B.
    const uint32_t drow = (uint32_t)(tid >> 2) * ROW_PITCH + (uint32_t)(tid & 3) * 16u;

    // ---- issue stage 0 ----
    {
        uint32_t dst = lds0 + drow;
        uint64_t src = gbase + (uint64_t)tid * 16u;
        asm volatile("global_load_async_to_lds_b128 %0, %1, off"
                     :: "v"(dst), "v"(src) : "memory");
        asm volatile("global_load_async_to_lds_b128 %0, %1, off"
                     :: "v"(dst + (uint32_t)(64 * ROW_PITCH)), "v"(src + 4096u) : "memory");
    }

    for (int s = 0; s < N_STAGES; ++s) {
        if (s + 1 < N_STAGES) {
            uint32_t dst = lds0 + (uint32_t)(((s + 1) & 1) * BUF_BYTES) + drow;
            uint64_t src = gbase + (uint64_t)(s + 1) * CHUNK_BYTES + (uint64_t)tid * 16u;
            asm volatile("global_load_async_to_lds_b128 %0, %1, off"
                         :: "v"(dst), "v"(src) : "memory");
            asm volatile("global_load_async_to_lds_b128 %0, %1, off"
                         :: "v"(dst + (uint32_t)(64 * ROW_PITCH)), "v"(src + 4096u) : "memory");
            asm volatile("s_wait_asynccnt 0x2" ::: "memory");   // stage s complete (in-order)
        } else {
            asm volatile("s_wait_asynccnt 0x0" ::: "memory");
        }
        __syncthreads();

        const char* eb = &ebuf[s & 1][0];
        #pragma unroll
        for (int tt = 0; tt < CHUNK_CODES / 16; ++tt) {
            const char* er = eb + (size_t)(tt * 16 + lo) * ROW_PITCH + hi * 32;
            v8bf blo = *(const v8bf*)er;
            v8bf bh  = *(const v8bf*)(er + 16);
            v16bf bm;
            #pragma unroll
            for (int i = 0; i < 8; ++i) { bm[i] = blo[i]; bm[i + 8] = bh[i]; }

            v8f sc = __builtin_amdgcn_wmma_f32_16x16x32_bf16(false, a, false, bm,
                                                             (short)0, zacc, false, false);
            const int code = s * CHUNK_CODES + tt * 16 + lo;
            #pragma unroll
            for (int j = 0; j < 8; ++j)
                if (sc[j] > best[j]) { best[j] = sc[j]; bidx[j] = code; }  // strict >: first-index ties
        }
        __syncthreads();   // buffer (s&1) may be overwritten by stage s+2's issue
    }

    // reduce across the 16 lanes sharing each pixel row (xor<=8 stays in half)
    #pragma unroll
    for (int off = 8; off >= 1; off >>= 1) {
        #pragma unroll
        for (int j = 0; j < 8; ++j) {
            float ov = __shfl_xor(best[j], off, 32);
            int   oi = __shfl_xor(bidx[j], off, 32);
            if (ov > best[j] || (ov == best[j] && oi < bidx[j])) { best[j] = ov; bidx[j] = oi; }
        }
    }
    if (lo == 0) {
        #pragma unroll
        for (int j = 0; j < 8; ++j)
            idxout[pixbase + j + 8 * hi] = bidx[j];
    }
}

// ---------------------------------------------------------------------------
// Gather z_q (transposed out), qloss partial, EMA scatter-adds
// ---------------------------------------------------------------------------
__global__ __launch_bounds__(256) void vq_gather(const int* __restrict__ idx,
                                                 const float* __restrict__ zf,
                                                 const float* __restrict__ emb,
                                                 float* __restrict__ out_zq,
                                                 float* __restrict__ out_idx,
                                                 float* __restrict__ counts,
                                                 float* __restrict__ dw,
                                                 float* __restrict__ qacc) {
    int n = blockIdx.x * 256 + threadIdx.x;
    if (n >= N_PIX) return;
    const int id = idx[n];
    const int b = n >> 12, p = n & (HW - 1);
    const float* er = emb + (size_t)id * E_DIM;
    const float* zr = zf  + (size_t)n  * E_DIM;
    float* dwr = dw + (size_t)id * E_DIM;
    float* oz  = out_zq + (size_t)b * E_DIM * HW + p;
    float sq = 0.f;
    #pragma unroll
    for (int e = 0; e < E_DIM; ++e) {
        float q = er[e], zz = zr[e];
        float d = q - zz;
        sq += d * d;
        oz[(size_t)e * HW] = q;               // [B,E,H,W] layout
        atomicAdd(&dwr[e], zz);
    }
    atomicAdd(&counts[id], 1.0f);
    out_idx[n] = (float)id;
    atomicAdd(qacc, sq);
}

__global__ __launch_bounds__(256) void vq_ema_a(const float* __restrict__ ema_cs,
                                                const float* __restrict__ counts,
                                                float* __restrict__ out_ncs,
                                                float* __restrict__ nacc) {
    int k = blockIdx.x * 256 + threadIdx.x;   // 8192
    float v = 0.99f * ema_cs[k] + 0.01f * counts[k];
    out_ncs[k] = v;
    atomicAdd(nacc, v);
}

__global__ __launch_bounds__(256) void vq_ema_b(const float* __restrict__ ema_w,
                                                const float* __restrict__ dw,
                                                const float* __restrict__ ncs,
                                                const float* __restrict__ nacc,
                                                const float* __restrict__ qacc,
                                                float* __restrict__ out_nw,
                                                float* __restrict__ out_nemb,
                                                float* __restrict__ out_qloss) {
    int i = blockIdx.x * 256 + threadIdx.x;   // 262144 = K*E
    int k = i >> 5;
    float w = 0.99f * ema_w[i] + 0.01f * dw[i];
    out_nw[i] = w;
    float n  = *nacc;
    float cs = ncs[k];
    float csm = (cs + 1e-5f) / (n + (float)K_DIM * 1e-5f) * n;
    out_nemb[i] = w / csm;
    if (i == 0) out_qloss[0] = 0.25f * qacc[0] / (float)(N_PIX * E_DIM);
}

// ---------------------------------------------------------------------------
extern "C" void kernel_launch(void* const* d_in, const int* in_sizes, int n_in,
                              void* d_out, int out_size, void* d_ws, size_t ws_size,
                              hipStream_t stream) {
    const float* x      = (const float*)d_in[0];
    const float* Wq     = (const float*)d_in[1];
    const float* bq     = (const float*)d_in[2];
    const float* emb    = (const float*)d_in[3];
    const float* ema_cs = (const float*)d_in[4];
    const float* ema_w  = (const float*)d_in[5];

    float* out       = (float*)d_out;
    float* out_zq    = out;                 // 1048576
    float* out_idx   = out + 1048576;       // 32768
    float* out_qloss = out + 1081344;       // 1
    float* out_nemb  = out + 1081345;       // 262144
    float* out_ncs   = out + 1343489;       // 8192
    float* out_nw    = out + 1351681;       // 262144

    char* ws = (char*)d_ws;
    float*  zf     = (float*)(ws + 0);              // 4 MB  z f32 [N,32]
    __bf16* zbuf   = (__bf16*)(ws + 4194304);       // 2 MB  z bf16
    __bf16* enb    = (__bf16*)(ws + 6291456);       // 512 KB normalized emb bf16
    __bf16* wqb    = (__bf16*)(ws + 6815744);       // 16 KB Wq bf16
    int*    idxb   = (int*)  (ws + 6832128);        // 128 KB indices
    float*  counts = (float*)(ws + 6963200);        // 32 KB
    float*  dw     = (float*)(ws + 6995968);        // 1 MB
    float*  qacc   = (float*)(ws + 8044544);        // 4 B
    float*  nacc   = (float*)(ws + 8044548);        // 4 B

    // zero counts+dw+qacc+nacc (contiguous)
    hipMemsetAsync(ws + 6963200, 0, 1048576 + 32768 + 8, stream);

    vq_prep_norm<<<K_DIM / 8, 256, 0, stream>>>(emb, enb);
    vq_prep_wq  <<<32, 256, 0, stream>>>(Wq, wqb);
    vq_conv     <<<N_PIX / 64, 128, 0, stream>>>(x, bq, wqb, zf, zbuf);
    vq_search   <<<N_PIX / 128, 256, 0, stream>>>(zbuf, enb, idxb);
    vq_gather   <<<N_PIX / 256, 256, 0, stream>>>(idxb, zf, emb, out_zq, out_idx,
                                                  counts, dw, qacc);
    vq_ema_a    <<<K_DIM / 256, 256, 0, stream>>>(ema_cs, counts, out_ncs, nacc);
    vq_ema_b    <<<(K_DIM * E_DIM) / 256, 256, 0, stream>>>(ema_w, dw, out_ncs, nacc,
                                                            qacc, out_nw, out_nemb, out_qloss);
}
